// BiModalAttention_42606075576768
// MI455X (gfx1250) — matrix operand order repaired and checked by reading, not compiled, not run on hardware
//
#include <hip/hip_runtime.h>

// ---- plain ext-vector types (trivial, union-safe) ----
typedef __attribute__((ext_vector_type(16))) __bf16 v16bf;
typedef __attribute__((ext_vector_type(8)))  __bf16 v8bf;
typedef __attribute__((ext_vector_type(4)))  __bf16 v4bf;
typedef __attribute__((ext_vector_type(2)))  __bf16 v2bf;
typedef __attribute__((ext_vector_type(8)))  float  v8f;
typedef __attribute__((ext_vector_type(4)))  float  f4;
typedef __attribute__((ext_vector_type(4)))  unsigned u32x4;
typedef __attribute__((ext_vector_type(8)))  unsigned u32x8;

union Frag16 { v8bf h[2]; v16bf v; };

#define T_SEQ  2048
#define D_DIM  512
#define BATCH  8
#define NKB    (T_SEQ/32)
#define LSTR   520   // bf16 row stride Qs/Ks: 1024B data + 16B TDM pad
#define VTSTR  40    // bf16 row stride Vt[d][key]: 64B data + 16B TDM pad
#define SSTR   36
#define PSTR   40

// dynamic-LDS byte offsets (dynamic LDS base == 0: usable directly as TDM lds_addr)
#define OFF_QS   0
#define OFF_K0   (OFF_QS + 32*LSTR*2)
#define OFF_K1   (OFF_K0 + 32*LSTR*2)
#define OFF_SS   (OFF_K1 + 32*LSTR*2)
#define OFF_PS   (OFF_SS + 32*SSTR*4)
#define OFF_AL   (OFF_PS + 32*PSTR*2)
#define OFF_LI   (OFF_AL + 32*4)
#define OFF_FL   (OFF_LI + 32*4)
#define OFF_V10  (OFF_FL + 16)
#define OFF_V11  (OFF_V10 + D_DIM*VTSTR*2)
#define SMEM0    (OFF_V11 + D_DIM*VTSTR*2)   // 189200 B
#define OFF_V20  SMEM0
#define OFF_V21  (OFF_V20 + D_DIM*VTSTR*2)
#define SMEM1    (OFF_V21 + D_DIM*VTSTR*2)   // 271120 B (<= 320KB WG limit)

// ---- TDM: 2D tensor_load_to_lds, descriptor per CDNA5 ISA ch.8 ----
// pad_interval code: 2^(code+1) DWORDs between pads; pad_amount code: (code+1) DWORDs pad.
__device__ __forceinline__ void tdm_load_2d(unsigned lds_addr, const void* gaddr,
                                            unsigned tensor_d0, unsigned tensor_d1,
                                            unsigned tile_d0, unsigned tile_d1,
                                            unsigned stride_d0,
                                            unsigned pad_interval, unsigned pad_amount)
{
    unsigned long long ga = (unsigned long long)gaddr;
    u32x4 g0;
    g0[0] = 1u;                                        // count=1 valid descriptor
    g0[1] = lds_addr;
    g0[2] = (unsigned)(ga & 0xffffffffu);
    g0[3] = (unsigned)((ga >> 32) & 0x01ffffffu) | 0x80000000u;  // addr[56:32] | type=2
    u32x8 g1;
    g1[0] = (1u << 16) | (1u << 20) | (pad_interval << 22) | (pad_amount << 25);
    g1[1] = (tensor_d0 & 0xffffu) << 16;
    g1[2] = ((tensor_d0 >> 16) & 0xffffu) | ((tensor_d1 & 0xffffu) << 16);
    g1[3] = ((tensor_d1 >> 16) & 0xffffu) | ((tile_d0 & 0xffffu) << 16);
    g1[4] = (tile_d1 & 0xffffu);
    g1[5] = stride_d0;
    g1[6] = 0u;
    g1[7] = 0u;
    asm volatile("tensor_load_to_lds %0, %1" :: "s"(g0), "s"(g1) : "memory");
}

// ---- one-time fp32 -> bf16 convert, row-major + transposed copies ----
__global__ void convert_tr(const float* __restrict__ src, void* __restrict__ dstRv,
                           void* __restrict__ dstTv)
{
    __shared__ __bf16 tile[32][36];
    __bf16* dstR = (__bf16*)dstRv;
    __bf16* dstT = (__bf16*)dstTv;
    const int b  = blockIdx.z;
    const int t0 = blockIdx.y * 32;
    const int d0 = blockIdx.x * 32;
    const int r  = threadIdx.x >> 3;
    const int c4 = (threadIdx.x & 7) * 4;

    f4 v = *(const f4*)(src + ((size_t)b*T_SEQ + t0 + r)*D_DIM + d0 + c4);
    v4bf q; q[0]=(__bf16)v[0]; q[1]=(__bf16)v[1]; q[2]=(__bf16)v[2]; q[3]=(__bf16)v[3];
    *(v4bf*)(dstR + ((size_t)b*T_SEQ + t0 + r)*D_DIM + d0 + c4) = q;
    *(v4bf*)&tile[r][c4] = q;
    __syncthreads();
    v4bf w;
    #pragma unroll
    for (int i = 0; i < 4; ++i) w[i] = tile[c4 + i][r];
    *(v4bf*)(dstT + ((size_t)b*D_DIM + d0 + r)*T_SEQ + t0 + c4) = w;
}

// ---- one key-block step, buffer parity fixed at compile time ----
template<int MODE, int PAR>
__device__ __forceinline__ void attn_step(
    int kb, char* smem,
    const __bf16* __restrict__ Kb, const __bf16* __restrict__ VT1,
    const __bf16* __restrict__ VT2, const unsigned char* __restrict__ Mb,
    v8f (&o1)[2][8], v8f (&o2)[MODE ? 2 : 1][MODE ? 8 : 1],
    float& mrow, float& lrow,
    const int wid, const int lane, const int ln, const int hf,
    const int m, const int n)
{
    constexpr unsigned KOFF  = PAR ? OFF_K1  : OFF_K0;
    constexpr unsigned V1OFF = PAR ? OFF_V11 : OFF_V10;
    constexpr unsigned V2OFF = PAR ? OFF_V21 : OFF_V20;
    constexpr unsigned KNXT  = PAR ? OFF_K0  : OFF_K1;
    constexpr unsigned V1NXT = PAR ? OFF_V10 : OFF_V11;
    constexpr unsigned V2NXT = PAR ? OFF_V20 : OFF_V21;

    __bf16* Qs  = (__bf16*)(smem + OFF_QS);
    __bf16* Ksb = (__bf16*)(smem + KOFF);
    __bf16* Vb1 = (__bf16*)(smem + V1OFF);
    __bf16* Vb2 = (__bf16*)(smem + V2OFF);
    float*  Ss  = (float*) (smem + OFF_SS);
    __bf16* Ps  = (__bf16*)(smem + OFF_PS);
    float*  Al  = (float*) (smem + OFF_AL);
    int*    Fl  = (int*)   (smem + OFF_FL);

    const v8f zero8 = {0.f,0.f,0.f,0.f,0.f,0.f,0.f,0.f};

    __syncthreads();   // consumers of the opposite-parity buffers are done
    if (wid == 0) {
        if (kb + 1 < NKB) {
            // prefetch next block into opposite parity; overlaps this block's compute
            tdm_load_2d(KNXT,  Kb  + (size_t)(kb+1)*32*D_DIM, D_DIM, 32, D_DIM, 32, D_DIM, 7, 3);
            tdm_load_2d(V1NXT, VT1 + (size_t)(kb+1)*32, T_SEQ, D_DIM, 32, D_DIM, T_SEQ, 3, 3);
            if (MODE == 1)
                tdm_load_2d(V2NXT, VT2 + (size_t)(kb+1)*32, T_SEQ, D_DIM, 32, D_DIM, T_SEQ, 3, 3);
            __builtin_amdgcn_s_wait_tensorcnt(MODE == 1 ? 3 : 2);  // this block landed
        } else {
            __builtin_amdgcn_s_wait_tensorcnt(0);
        }
    }
    __syncthreads();   // this block's tiles visible to all waves

    // ---- S tile (m,n) = Q K^T over full D: 16 WMMAs ----
    v8f s = zero8;
    const __bf16* qrow = Qs  + (m*16 + ln)*LSTR;
    const __bf16* krow = Ksb + (n*16 + ln)*LSTR;
    #pragma unroll
    for (int c = 0; c < D_DIM/32; ++c) {
        Frag16 A, Bk;
        A.h[0]  = *(const v8bf*)(qrow + c*32 + hf*16);
        A.h[1]  = *(const v8bf*)(qrow + c*32 + hf*16 + 8);
        Bk.h[0] = *(const v8bf*)(krow + c*32 + hf*16);
        Bk.h[1] = *(const v8bf*)(krow + c*32 + hf*16 + 8);
        s = __builtin_amdgcn_wmma_f32_16x16x32_bf16(false, A.v, false, Bk.v,
                                                    (short)0, s, false, false);
    }
    #pragma unroll
    for (int k = 0; k < 8; ++k)
        Ss[(m*16 + hf*8 + k)*SSTR + n*16 + ln] = s[k];
    __syncthreads();

    // ---- online softmax: wave 0, one row per lane ----
    if (wid == 0) {
        float sv[32];
        float mx = -__builtin_inff();
        #pragma unroll
        for (int c4 = 0; c4 < 8; ++c4) {
            f4 v4 = *(const f4*)(Ss + lane*SSTR + c4*4);
            unsigned mk = *(const unsigned*)(Mb + kb*32 + c4*4);
            #pragma unroll
            for (int dd = 0; dd < 4; ++dd) {
                float v = ((mk >> (8*dd)) & 0xffu) ? v4[dd] : -10000.0f;
                sv[c4*4 + dd] = v;
                mx = fmaxf(mx, v);
            }
        }
        float mnew = fmaxf(mrow, mx);
        float al   = __expf(mrow - mnew);
        float sum  = 0.f;
        #pragma unroll
        for (int c2 = 0; c2 < 16; ++c2) {
            float p0 = __expf(sv[2*c2]   - mnew);
            float p1 = __expf(sv[2*c2+1] - mnew);
            sum += p0 + p1;
            v2bf pp; pp[0] = (__bf16)p0; pp[1] = (__bf16)p1;
            *(v2bf*)(Ps + lane*PSTR + 2*c2) = pp;
        }
        lrow = lrow * al + sum;
        mrow = mnew;
        Al[lane] = al;
        unsigned long long bal = __ballot(al != 1.0f);
        if (lane == 0) Fl[0] = (bal != 0ull) ? 1 : 0;
    }
    __syncthreads();

    // ---- rescale O only when a row max moved (block-uniform) ----
    if (Fl[0]) {
        #pragma unroll
        for (int mi = 0; mi < 2; ++mi)
            #pragma unroll
            for (int k = 0; k < 8; ++k) {
                float al = Al[mi*16 + hf*8 + k];
                #pragma unroll
                for (int nd = 0; nd < 8; ++nd) {
                    o1[mi][nd][k] *= al;
                    if (MODE == 1) o2[mi][nd][k] *= al;
                }
            }
    }
    // ---- O += P @ V (and P @ V2) ----
    Frag16 P0, P1;
    P0.h[0] = *(const v8bf*)(Ps + (0*16 + ln)*PSTR + hf*16);
    P0.h[1] = *(const v8bf*)(Ps + (0*16 + ln)*PSTR + hf*16 + 8);
    P1.h[0] = *(const v8bf*)(Ps + (1*16 + ln)*PSTR + hf*16);
    P1.h[1] = *(const v8bf*)(Ps + (1*16 + ln)*PSTR + hf*16 + 8);
    #pragma unroll
    for (int nd = 0; nd < 8; ++nd) {
        int col = wid*128 + nd*16 + ln;
        Frag16 Bv;
        Bv.h[0] = *(const v8bf*)(Vb1 + col*VTSTR + hf*16);
        Bv.h[1] = *(const v8bf*)(Vb1 + col*VTSTR + hf*16 + 8);
        o1[0][nd] = __builtin_amdgcn_wmma_f32_16x16x32_bf16(false, P0.v, false, Bv.v,
                                                            (short)0, o1[0][nd], false, false);
        o1[1][nd] = __builtin_amdgcn_wmma_f32_16x16x32_bf16(false, P1.v, false, Bv.v,
                                                            (short)0, o1[1][nd], false, false);
        if (MODE == 1) {
            Frag16 Bw;
            Bw.h[0] = *(const v8bf*)(Vb2 + col*VTSTR + hf*16);
            Bw.h[1] = *(const v8bf*)(Vb2 + col*VTSTR + hf*16 + 8);
            o2[0][nd] = __builtin_amdgcn_wmma_f32_16x16x32_bf16(false, P0.v, false, Bw.v,
                                                                (short)0, o2[0][nd], false, false);
            o2[1][nd] = __builtin_amdgcn_wmma_f32_16x16x32_bf16(false, P1.v, false, Bw.v,
                                                                (short)0, o2[1][nd], false, false);
        }
    }
}

// MODE 0 (phase 1): out = transposed bf16 oyT[b][d][t], gated
// MODE 1 (phase 2): dual-V; O1 gated -> out[:,:512] f32, O2 -> out[:,512:] f32
template<int MODE>
__global__ __launch_bounds__(128, 1)
void flash_attn(const void* __restrict__ Qg, const void* __restrict__ Kg,
                const void* __restrict__ VT1g, const void* __restrict__ VT2g,
                const unsigned char* __restrict__ maskg,
                const float* __restrict__ gateg, void* __restrict__ outg)
{
    extern __shared__ char smem[];

    const int tid  = threadIdx.x;
    const int wid  = tid >> 5;
    const int lane = tid & 31;
    const int ln   = lane & 15;
    const int hf   = lane >> 4;
    const int m    = wid >> 1;
    const int n    = wid & 1;

    const int b  = blockIdx.y;
    const int qb = blockIdx.x;

    const __bf16* Qb  = (const __bf16*)Qg   + ((size_t)b*T_SEQ + qb*32) * D_DIM;
    const __bf16* Kb  = (const __bf16*)Kg   + (size_t)b*T_SEQ*D_DIM;
    const __bf16* VT1 = (const __bf16*)VT1g + (size_t)b*D_DIM*T_SEQ;  // [D][T]
    const __bf16* VT2 = (const __bf16*)VT2g + (size_t)b*D_DIM*T_SEQ;  // [D][T]
    const unsigned char* Mb = maskg + (size_t)b*T_SEQ;
    const float*  Gb  = gateg + ((size_t)b*T_SEQ + qb*32) * D_DIM;

    // prologue: Q tile + key-block 0 in flight before the loop
    if (wid == 0) {
        tdm_load_2d(OFF_QS, Qb, D_DIM, 32, D_DIM, 32, D_DIM, 7, 3);
        tdm_load_2d(OFF_K0, Kb, D_DIM, 32, D_DIM, 32, D_DIM, 7, 3);
        tdm_load_2d(OFF_V10, VT1, T_SEQ, D_DIM, 32, D_DIM, T_SEQ, 3, 3);
        if (MODE == 1)
            tdm_load_2d(OFF_V20, VT2, T_SEQ, D_DIM, 32, D_DIM, T_SEQ, 3, 3);
    }

    const v8f zero8 = {0.f,0.f,0.f,0.f,0.f,0.f,0.f,0.f};
    v8f o1[2][8];
    v8f o2[MODE ? 2 : 1][MODE ? 8 : 1];
    #pragma unroll
    for (int a = 0; a < 2; ++a)
        #pragma unroll
        for (int c = 0; c < 8; ++c) {
            o1[a][c] = zero8;
            if (MODE == 1) o2[a][c] = zero8;
        }

    float mrow = -__builtin_inff();
    float lrow = 0.f;

    for (int kb = 0; kb < NKB; kb += 2) {   // NKB even: parity fixed per sub-step
        attn_step<MODE, 0>(kb,     smem, Kb, VT1, VT2, Mb, o1, o2, mrow, lrow,
                           wid, lane, ln, hf, m, n);
        attn_step<MODE, 1>(kb + 1, smem, Kb, VT1, VT2, Mb, o1, o2, mrow, lrow,
                           wid, lane, ln, hf, m, n);
    }

    float* Li = (float*)(smem + OFF_LI);
    if (wid == 0) Li[lane] = 1.0f / lrow;
    __syncthreads();

    // ---- epilogue ----
    #pragma unroll
    for (int mi = 0; mi < 2; ++mi)
        #pragma unroll
        for (int k = 0; k < 8; ++k) {
            int rl = mi*16 + hf*8 + k;
            float li = Li[rl];
            #pragma unroll
            for (int nd = 0; nd < 8; ++nd) {
                int col = wid*128 + nd*16 + ln;
                if (MODE == 0) {
                    // write output_y transposed: oyT[b][d=col][t=row]
                    float val = o1[mi][nd][k] * li * Gb[rl*D_DIM + col];
                    ((__bf16*)outg)[((size_t)b*D_DIM + col)*T_SEQ + qb*32 + rl] = (__bf16)val;
                } else {
                    size_t gr = (size_t)b*T_SEQ + qb*32 + rl;
                    float va = o1[mi][nd][k] * li * Gb[rl*D_DIM + col];
                    float vb = o2[mi][nd][k] * li;
                    ((float*)outg)[gr*(size_t)(2*D_DIM) + col]         = va;
                    ((float*)outg)[gr*(size_t)(2*D_DIM) + D_DIM + col] = vb;
                }
            }
        }
}

extern "C" void kernel_launch(void* const* d_in, const int* in_sizes, int n_in,
                              void* d_out, int out_size, void* d_ws, size_t ws_size,
                              hipStream_t stream) {
    (void)in_sizes; (void)n_in; (void)out_size; (void)ws_size;
    const float* x = (const float*)d_in[0];
    const float* y = (const float*)d_in[1];
    const unsigned char* xm = (const unsigned char*)d_in[2];
    const unsigned char* ym = (const unsigned char*)d_in[3];

    char* w = (char*)d_ws;                                   // ws layout (bf16 tensors):
    const size_t NTDB = (size_t)BATCH*T_SEQ*D_DIM*2;         // 16.8 MB each
    void* xb  = w;              // x bf16 row-major
    void* yb  = w + 1*NTDB;     // y bf16 row-major
    void* xbT = w + 2*NTDB;     // x bf16 [B][D][T]
    void* ybT = w + 3*NTDB;     // y bf16 [B][D][T]
    void* oyT = w + 4*NTDB;     // output_y bf16 [B][D][T]

    dim3 cg(D_DIM/32, T_SEQ/32, BATCH);
    convert_tr<<<cg, 256, 0, stream>>>(x, xb, xbT);
    convert_tr<<<cg, 256, 0, stream>>>(y, yb, ybT);

    dim3 grid(T_SEQ/32, BATCH);
    // phase 1: output_y = (softmax(y x^T | x_mask) @ x) * y  -> oyT (transposed bf16)
    flash_attn<0><<<grid, 128, SMEM0, stream>>>(yb, xb, xbT, xbT, xm, y, oyT);
    // phase 2 (fused): O1 -> d_out[:,:512], O2 -> d_out[:,512:]
    flash_attn<1><<<grid, 128, SMEM1, stream>>>(xb, yb, ybT, oyT, ym, x, d_out);
}
